// MultiGraph_90881507983368
// MI455X (gfx1250) — compile-verified, matrix operand order checked by reading, not run on hardware
//
#include <hip/hip_runtime.h>
#include <stdint.h>

// ---------------- problem constants ----------------
#define N_NODES   20000
#define E_EDGES   640000
#define BATCH     256
#define HEADS     4
#define DIM       64
#define HD        (HEADS * DIM)          // 256
#define EE_EDGES  (E_EDGES + N_NODES)    // supra edges + self loops
#define NEG_SLOPE 0.2f
#define LN_EPS    1e-5f

typedef unsigned short ushort_t;         // bf16 storage
typedef __attribute__((ext_vector_type(16))) __bf16 v16bf;
typedef __attribute__((ext_vector_type(8)))  float  v8f;

union Frag { uint4 u[2]; v16bf v; };

__device__ __forceinline__ ushort_t f2bf(float f) {
  unsigned int u = __float_as_uint(f);
  unsigned int r = u + 0x7FFFu + ((u >> 16) & 1u);   // round-to-nearest-even
  return (ushort_t)(r >> 16);
}

__device__ __forceinline__ void atomicMaxF(float* addr, float val) {
  unsigned int* ua = (unsigned int*)addr;
  unsigned int old = __float_as_uint(*addr);
  while (__uint_as_float(old) < val) {
    unsigned int assumed = old;
    old = atomicCAS(ua, assumed, __float_as_uint(val));
    if (old == assumed) break;
  }
}

__device__ __forceinline__ float lrelu(float v) {
  return v > 0.f ? v : NEG_SLOPE * v;
}

// ---------------- utility kernels ----------------
__global__ void k_fill_f32(float* p, float v, int n) {
  int i = blockIdx.x * blockDim.x + threadIdx.x;
  if (i < n) p[i] = v;
}
__global__ void k_fill_i32(int* p, int v, int n) {
  int i = blockIdx.x * blockDim.x + threadIdx.x;
  if (i < n) p[i] = v;
}
__global__ void k_f32_to_bf16(const float* __restrict__ src, ushort_t* __restrict__ dst, int n) {
  int i = blockIdx.x * blockDim.x + threadIdx.x;
  if (i < n) dst[i] = f2bf(src[i]);
}
// src [rows, DIM] f32 -> dst bf16 rows with arbitrary stride/col offset (for concat build)
__global__ void k_conv_strided(const float* __restrict__ src, ushort_t* __restrict__ dst,
                               int rows, int dstStride, int colOff) {
  int i = blockIdx.x * blockDim.x + threadIdx.x;
  if (i < rows * DIM) {
    int r = i >> 6, d = i & 63;
    dst[(size_t)r * dstStride + colOff + d] = f2bf(src[i]);
  }
}
// transpose-convert weight: src [R,C] f32 -> dst [C,R] bf16
__global__ void k_tconv(const float* __restrict__ src, ushort_t* __restrict__ dst, int R, int C) {
  int i = blockIdx.x * blockDim.x + threadIdx.x;
  if (i < R * C) {
    int r = i / C, c = i % C;
    dst[(size_t)c * R + r] = f2bf(src[i]);
  }
}

// ---- WMMA GEMM: C[M,Nc] = act(A[M,K](bf16) @ Bt[Nc,K]^T(bf16) + bias) ----
// One wave computes a 16x64 output tile: 1 shared A fragment x 4 B fragments
// per K-step -> 4 accumulators, K templated so the loop fully unrolls.
// grid = (Nc/64, M/16), block = 32. K % 32 == 0, Nc % 64 == 0.
// 16-bit A-fragment layout (CDNA5 ISA): lanes 0-15 row M=lane, K {kb+0..7, kb+16..23};
// lanes 16-31 row M=lane-16, K {kb+8..15, kb+24..31}. Bt pre-transposed so B loads
// use the identical pattern with lane = output column.
template <int K>
__global__ __launch_bounds__(32) void k_wmma_gemm4(const ushort_t* __restrict__ A,
                                                   const ushort_t* __restrict__ Bt,
                                                   const float* __restrict__ bias,
                                                   float* __restrict__ C,
                                                   int Nc, int act) {
  int lane = threadIdx.x & 31;
  int tn = blockIdx.x, tm = blockIdx.y;
  int half = lane >> 4, l15 = lane & 15;
  const ushort_t* ap = A + (size_t)(tm * 16 + l15) * K + half * 8;
  const ushort_t* bp = Bt + (size_t)(tn * 64 + l15) * K + half * 8;
  v8f cc[4];
#pragma unroll
  for (int j = 0; j < 4; ++j) cc[j] = (v8f){0.f, 0.f, 0.f, 0.f, 0.f, 0.f, 0.f, 0.f};
#pragma unroll
  for (int kb = 0; kb < K; kb += 32) {
    Frag a;
    a.u[0] = *(const uint4*)(ap + kb);
    a.u[1] = *(const uint4*)(ap + kb + 16);
#pragma unroll
    for (int j = 0; j < 4; ++j) {
      Frag b;
      const ushort_t* bpj = bp + (size_t)j * 16 * K;
      b.u[0] = *(const uint4*)(bpj + kb);
      b.u[1] = *(const uint4*)(bpj + kb + 16);
      cc[j] = __builtin_amdgcn_wmma_f32_16x16x32_bf16(false, a.v, false, b.v,
                                                      (short)0, cc[j], false, false);
    }
  }
  // C/D layout: VGPR v -> row tm*16 + v + (lane>=16 ? 8 : 0), col = tile_col + (lane&15)
  int rbase = tm * 16 + half * 8;
#pragma unroll
  for (int j = 0; j < 4; ++j) {
    int col = tn * 64 + j * 16 + l15;
    float bv = bias ? bias[col] : 0.f;
#pragma unroll
    for (int v = 0; v < 8; ++v) {
      float x = cc[j][v] + bv;
      if (act) x = fmaxf(x, 0.f);
      C[(size_t)(rbase + v) * Nc + col] = x;
    }
  }
}

// ---------------- GATv2 edge kernels (one wave per edge, float4 gathers) ----------------
// 256 channels = 2 steps x 32 lanes x float4; head = 2*st + (lane>>4)
__global__ void k_edge_pass1(const int* __restrict__ ei, const float* __restrict__ xl,
                             const float* __restrict__ xr, const float* __restrict__ att,
                             float* __restrict__ estore, float* __restrict__ emax) {
  int wv = (blockIdx.x * blockDim.x + threadIdx.x) >> 5;
  int lane = threadIdx.x & 31;
  if (wv >= EE_EDGES) return;
  int s, d;
  if (wv < E_EDGES) { s = ei[wv]; d = ei[E_EDGES + wv]; } else { s = d = wv - E_EDGES; }
  const float4* pl4 = (const float4*)(xl + (size_t)s * HD);
  const float4* pr4 = (const float4*)(xr + (size_t)d * HD);
  const float4* at4 = (const float4*)att;
  int half = lane >> 4;
  float acc0 = 0.f, acc1 = 0.f, acc2 = 0.f, acc3 = 0.f;
#pragma unroll
  for (int st = 0; st < 2; ++st) {
    int i4 = st * 32 + lane;
    float4 a = pl4[i4], b = pr4[i4], w = at4[i4];
    float t = lrelu(a.x + b.x) * w.x + lrelu(a.y + b.y) * w.y +
              lrelu(a.z + b.z) * w.z + lrelu(a.w + b.w) * w.w;
    if (st == 0) { if (half) acc1 += t; else acc0 += t; }
    else         { if (half) acc3 += t; else acc2 += t; }
  }
#pragma unroll
  for (int m = 16; m >= 1; m >>= 1) {
    acc0 += __shfl_xor(acc0, m, 32);
    acc1 += __shfl_xor(acc1, m, 32);
    acc2 += __shfl_xor(acc2, m, 32);
    acc3 += __shfl_xor(acc3, m, 32);
  }
  if (lane < 4) {
    float mine = lane == 0 ? acc0 : lane == 1 ? acc1 : lane == 2 ? acc2 : acc3;
    estore[(size_t)wv * 4 + lane] = mine;
    atomicMaxF(&emax[(size_t)d * 4 + lane], mine);
  }
}

__global__ void k_edge_pass2(const int* __restrict__ ei, const float* __restrict__ xl,
                             const float* __restrict__ estore, const float* __restrict__ emax,
                             float* __restrict__ denom, float* __restrict__ numer) {
  int wv = (blockIdx.x * blockDim.x + threadIdx.x) >> 5;
  int lane = threadIdx.x & 31;
  if (wv >= EE_EDGES) return;
  int s, d;
  if (wv < E_EDGES) { s = ei[wv]; d = ei[E_EDGES + wv]; } else { s = d = wv - E_EDGES; }
  const float4* pl4 = (const float4*)(xl + (size_t)s * HD);
  float* np = numer + (size_t)d * HD;
  float4 ev = *(const float4*)(estore + (size_t)wv * 4);
  float4 mv = *(const float4*)(emax + (size_t)d * 4);
  float x0 = __expf(ev.x - mv.x), x1 = __expf(ev.y - mv.y);
  float x2 = __expf(ev.z - mv.z), x3 = __expf(ev.w - mv.w);
  if (lane < 4) {
    float mine = lane == 0 ? x0 : lane == 1 ? x1 : lane == 2 ? x2 : x3;
    atomicAdd(&denom[(size_t)d * 4 + lane], mine);
  }
  int half = lane >> 4;
#pragma unroll
  for (int st = 0; st < 2; ++st) {
    int i4 = st * 32 + lane;
    float4 a = pl4[i4];
    float exh = (st == 0) ? (half ? x1 : x0) : (half ? x3 : x2);
    atomicAdd(&np[i4 * 4 + 0], exh * a.x);
    atomicAdd(&np[i4 * 4 + 1], exh * a.y);
    atomicAdd(&np[i4 * 4 + 2], exh * a.z);
    atomicAdd(&np[i4 * 4 + 3], exh * a.w);
  }
}

// h = relu(mean_h(numer/denom) + bias); write f32 and bf16 (next-layer input)
__global__ void k_node_finish(const float* __restrict__ numer, const float* __restrict__ denom,
                              const float* __restrict__ bias, float* __restrict__ hout,
                              ushort_t* __restrict__ hbf) {
  int i = blockIdx.x * blockDim.x + threadIdx.x;
  if (i >= N_NODES * DIM) return;
  int n = i >> 6, d = i & 63;
  const float* np = numer + (size_t)n * HD;
  const float* dp = denom + (size_t)n * 4;
  float acc = 0.f;
#pragma unroll
  for (int h = 0; h < 4; ++h) acc += np[h * DIM + d] / (dp[h] + 1e-16f);
  float v = fmaxf(acc * 0.25f + bias[d], 0.f);
  hout[i] = v;
  hbf[i] = f2bf(v);
}

// ---------------- struct branch ----------------
__global__ void k_hist(const int* __restrict__ dst, int* __restrict__ indeg) {
  int i = blockIdx.x * blockDim.x + threadIdx.x;
  if (i < E_EDGES) atomicAdd(&indeg[dst[i]], 1);
}
// ea = MLP2(1): constant per-edge vector (edge weights are all ones)
__global__ void k_ea(const float* fe1W, const float* fe1b, const float* fe2W,
                     const float* fe2b, float* ea) {
  int t = threadIdx.x;
  if (t >= DIM) return;
  float acc = fe2b[t];
  for (int k = 0; k < DIM; ++k) {
    float h = fmaxf(fe1W[k] + fe1b[k], 0.f);
    acc += h * fe2W[k * DIM + t];
  }
  ea[t] = acc;
}
__global__ void k_nspre(const int* __restrict__ indeg, const float* __restrict__ ea,
                        ushort_t* __restrict__ nspre) {
  int i = blockIdx.x * blockDim.x + threadIdx.x;
  if (i < N_NODES * DIM) {
    int n = i >> 6, d = i & 63;
    nspre[i] = f2bf((float)indeg[n] * ea[d]);
  }
}

// ---------------- gated combiner + LayerNorm (one wave per node) ----------------
__global__ __launch_bounds__(32) void k_combine(const float* __restrict__ zb,
                                                const float* __restrict__ stb,
                                                const float* __restrict__ gtb,
                                                const float* __restrict__ ns,
                                                const float* __restrict__ gnn,
                                                const float* __restrict__ ln_g,
                                                const float* __restrict__ ln_b,
                                                float* __restrict__ outbuf, int accum) {
  int n = blockIdx.x;
  int lane = threadIdx.x & 31;
  size_t base = (size_t)n * DIM;
  float v[2];
#pragma unroll
  for (int j = 0; j < 2; ++j) {
    int d = lane + 32 * j;
    float z = 1.f / (1.f + __expf(-zb[base + d]));
    v[j] = z * stb[base + d] + (1.f - z) * gtb[base + d] + ns[base + d] + gnn[base + d];
  }
  float s = v[0] + v[1];
#pragma unroll
  for (int m = 16; m >= 1; m >>= 1) s += __shfl_xor(s, m, 32);
  float mean = s * (1.f / 64.f);
  float q = (v[0] - mean) * (v[0] - mean) + (v[1] - mean) * (v[1] - mean);
#pragma unroll
  for (int m = 16; m >= 1; m >>= 1) q += __shfl_xor(q, m, 32);
  float inv = rsqrtf(q * (1.f / 64.f) + LN_EPS);
#pragma unroll
  for (int j = 0; j < 2; ++j) {
    int d = lane + 32 * j;
    float o = (v[j] - mean) * inv * ln_g[d] + ln_b[d];
    if (accum) outbuf[base + d] += 0.5f * o;
    else       outbuf[base + d]  = 0.5f * o;
  }
}

__global__ void k_gather(const float* __restrict__ outbuf, const int* __restrict__ p,
                         float* __restrict__ out) {
  int i = blockIdx.x * blockDim.x + threadIdx.x;
  if (i < BATCH * DIM) {
    int b = i >> 6, d = i & 63;
    out[i] = outbuf[(size_t)p[b] * DIM + d];
  }
}

// ---------------- host orchestration ----------------
static inline int g1(int n) { return (n + 255) / 256; }

extern "C" void kernel_launch(void* const* d_in, const int* in_sizes, int n_in,
                              void* d_out, int out_size, void* d_ws, size_t ws_size,
                              hipStream_t stream) {
  (void)in_sizes; (void)n_in; (void)out_size; (void)ws_size;

  // Input order: first 6 in setup_inputs() insertion order; params flattened as a
  // jax pytree (dict keys sorted): comb[0..1]{gateW,gateB,gtW,gtB,ln_b,ln_g,stW,stB},
  // features, gat[0..1]{Wl,Wr,att,bias}, struct[0..1]{fe1W,fe1b,fe2W,fe2b,fn1W,fn1b,fn2W,fn2b}
  const int IN_EI_G1 = 0, IN_EI_G2 = 2, IN_EI_SUPRA = 4, IN_P = 5;
  const int IN_COMB = 6;       // +8 per layer
  const int IN_FEATURES = 22;
  const int IN_GAT = 23;       // +4 per layer
  const int IN_STRUCT = 31;    // +8 per layer

  char* ws = (char*)d_ws;
  size_t o = 0;
  auto alloc = [&](size_t bytes) -> size_t {
    size_t r = (o + 255) & ~(size_t)255; o = r + bytes; return r;
  };
  const size_t ND = (size_t)N_NODES * DIM;
  const size_t NHD = (size_t)N_NODES * HD;

  size_t xbf_o   = alloc(ND * 2);                 // current-layer input bf16 (ends = gnn bf16)
  size_t hbuf_o  = alloc(ND * 4);                 // layer output f32 (ends = gnn f32)
  size_t outb_o  = alloc(ND * 4);                 // fused combiner output
  size_t wt_o    = alloc(1 << 17);                // transposed-weight pool (bf16)
  size_t ea_o    = alloc(256);
  size_t indeg_o = alloc((size_t)N_NODES * 4);
  size_t emax_o  = alloc((size_t)N_NODES * HEADS * 4);
  size_t den_o   = alloc((size_t)N_NODES * HEADS * 4);
  size_t big_o   = alloc(3 * NHD * 4 + (size_t)EE_EDGES * HEADS * 4);

  ushort_t* xbf   = (ushort_t*)(ws + xbf_o);
  float*    hbuf  = (float*)(ws + hbuf_o);
  float*    outb  = (float*)(ws + outb_o);
  float*    ea    = (float*)(ws + ea_o);
  int*      indeg = (int*)(ws + indeg_o);
  float*    emax  = (float*)(ws + emax_o);
  float*    den   = (float*)(ws + den_o);

  // GAT-phase view of the big region
  float* xl     = (float*)(ws + big_o);
  float* xr     = xl + NHD;
  float* numer  = xr + NHD;
  float* estore = numer + NHD;
  // struct/combiner-phase view (reuses big region after GAT completes)
  char* bg = ws + big_o;
  ushort_t* nspre = (ushort_t*)bg;                      // ND*2  = 2,560,000
  float*    t1    = (float*)(bg + 2560000);             // ND*4
  ushort_t* t1bf  = (ushort_t*)(bg + 7680000);          // ND*2
  float*    nsb   = (float*)(bg + 10240000);            // ND*4
  ushort_t* nsbf  = (ushort_t*)(bg + 15360000);         // ND*2
  ushort_t* cat   = (ushort_t*)(bg + 17920000);         // N*128*2
  float*    zbuf  = (float*)(bg + 23040000);
  float*    stb   = (float*)(bg + 28160000);
  float*    gtb   = (float*)(bg + 33280000);

  // weight pool slots (bf16 elements)
  ushort_t* wtA    = (ushort_t*)(ws + wt_o);   // WlT [256,64]
  ushort_t* wtB    = wtA + 16384;              // WrT [256,64]
  ushort_t* w_fn1  = wtA;                      // [64,64]
  ushort_t* w_fn2  = wtA + 4096;
  ushort_t* w_gate = wtA + 8192;               // [64,128]
  ushort_t* w_st   = wtA + 16384;
  ushort_t* w_gt   = wtA + 20480;

  const int* supra = (const int*)d_in[IN_EI_SUPRA];
  const int edge_threads = EE_EDGES * 32;

  // features -> bf16
  k_f32_to_bf16<<<g1(ND), 256, 0, stream>>>((const float*)d_in[IN_FEATURES], xbf, (int)ND);

  // ---- 2x GATv2 layers over supra graph ----
  for (int L = 0; L < 2; ++L) {
    const float* Wl    = (const float*)d_in[IN_GAT + 4 * L + 0];
    const float* Wr    = (const float*)d_in[IN_GAT + 4 * L + 1];
    const float* att   = (const float*)d_in[IN_GAT + 4 * L + 2];
    const float* gbias = (const float*)d_in[IN_GAT + 4 * L + 3];

    k_tconv<<<g1(DIM * HD), 256, 0, stream>>>(Wl, wtA, DIM, HD);
    k_tconv<<<g1(DIM * HD), 256, 0, stream>>>(Wr, wtB, DIM, HD);
    k_wmma_gemm4<DIM><<<dim3(HD / 64, N_NODES / 16), 32, 0, stream>>>(xbf, wtA, nullptr, xl, HD, 0);
    k_wmma_gemm4<DIM><<<dim3(HD / 64, N_NODES / 16), 32, 0, stream>>>(xbf, wtB, nullptr, xr, HD, 0);

    k_fill_f32<<<g1(N_NODES * HEADS), 256, 0, stream>>>(emax, -3.0e38f, N_NODES * HEADS);
    k_fill_f32<<<g1(N_NODES * HEADS), 256, 0, stream>>>(den, 0.f, N_NODES * HEADS);
    k_fill_f32<<<g1((int)NHD), 256, 0, stream>>>(numer, 0.f, (int)NHD);

    k_edge_pass1<<<g1(edge_threads), 256, 0, stream>>>(supra, xl, xr, att, estore, emax);
    k_edge_pass2<<<g1(edge_threads), 256, 0, stream>>>(supra, xl, estore, emax, den, numer);
    k_node_finish<<<g1((int)ND), 256, 0, stream>>>(numer, den, gbias, hbuf, xbf);
  }
  // hbuf = gnn_emb (f32), xbf = gnn_emb (bf16)

  // ---- 2x struct branch + gated combiner ----
  for (int l = 0; l < 2; ++l) {
    const int* ei = (const int*)d_in[l == 0 ? IN_EI_G1 : IN_EI_G2];
    const float* fe1W = (const float*)d_in[IN_STRUCT + 8 * l + 0];
    const float* fe1b = (const float*)d_in[IN_STRUCT + 8 * l + 1];
    const float* fe2W = (const float*)d_in[IN_STRUCT + 8 * l + 2];
    const float* fe2b = (const float*)d_in[IN_STRUCT + 8 * l + 3];
    const float* fn1W = (const float*)d_in[IN_STRUCT + 8 * l + 4];
    const float* fn1b = (const float*)d_in[IN_STRUCT + 8 * l + 5];
    const float* fn2W = (const float*)d_in[IN_STRUCT + 8 * l + 6];
    const float* fn2b = (const float*)d_in[IN_STRUCT + 8 * l + 7];
    const float* gateW = (const float*)d_in[IN_COMB + 8 * l + 0];
    const float* gateB = (const float*)d_in[IN_COMB + 8 * l + 1];
    const float* gtW   = (const float*)d_in[IN_COMB + 8 * l + 2];
    const float* gtB   = (const float*)d_in[IN_COMB + 8 * l + 3];
    const float* ln_b  = (const float*)d_in[IN_COMB + 8 * l + 4];
    const float* ln_g  = (const float*)d_in[IN_COMB + 8 * l + 5];
    const float* stW   = (const float*)d_in[IN_COMB + 8 * l + 6];
    const float* stB   = (const float*)d_in[IN_COMB + 8 * l + 7];

    // ns_pre[n] = indeg(n) * MLP(1)   (edge weights are all-ones)
    k_fill_i32<<<g1(N_NODES), 256, 0, stream>>>(indeg, 0, N_NODES);
    k_hist<<<g1(E_EDGES), 256, 0, stream>>>(ei + E_EDGES, indeg);
    k_ea<<<1, 64, 0, stream>>>(fe1W, fe1b, fe2W, fe2b, ea);
    k_nspre<<<g1((int)ND), 256, 0, stream>>>(indeg, ea, nspre);

    // ns = fn2(relu(fn1(ns_pre)))
    k_tconv<<<g1(DIM * DIM), 256, 0, stream>>>(fn1W, w_fn1, DIM, DIM);
    k_tconv<<<g1(DIM * DIM), 256, 0, stream>>>(fn2W, w_fn2, DIM, DIM);
    k_wmma_gemm4<DIM><<<dim3(1, N_NODES / 16), 32, 0, stream>>>(nspre, w_fn1, fn1b, t1, DIM, 1);
    k_f32_to_bf16<<<g1((int)ND), 256, 0, stream>>>(t1, t1bf, (int)ND);
    k_wmma_gemm4<DIM><<<dim3(1, N_NODES / 16), 32, 0, stream>>>(t1bf, w_fn2, fn2b, nsb, DIM, 0);

    // build bf16 operands: nsbf, cat = [ns | gnn]
    k_conv_strided<<<g1((int)ND), 256, 0, stream>>>(nsb, nsbf, N_NODES, DIM, 0);
    k_conv_strided<<<g1((int)ND), 256, 0, stream>>>(nsb, cat, N_NODES, 2 * DIM, 0);
    k_conv_strided<<<g1((int)ND), 256, 0, stream>>>(hbuf, cat, N_NODES, 2 * DIM, DIM);

    // gate / st / gt linear layers
    k_tconv<<<g1(2 * DIM * DIM), 256, 0, stream>>>(gateW, w_gate, 2 * DIM, DIM);
    k_tconv<<<g1(DIM * DIM), 256, 0, stream>>>(stW, w_st, DIM, DIM);
    k_tconv<<<g1(DIM * DIM), 256, 0, stream>>>(gtW, w_gt, DIM, DIM);
    k_wmma_gemm4<2 * DIM><<<dim3(1, N_NODES / 16), 32, 0, stream>>>(cat,  w_gate, gateB, zbuf, DIM, 0);
    k_wmma_gemm4<DIM><<<dim3(1, N_NODES / 16), 32, 0, stream>>>(nsbf, w_st, stB, stb, DIM, 0);
    k_wmma_gemm4<DIM><<<dim3(1, N_NODES / 16), 32, 0, stream>>>(xbf,  w_gt, gtB, gtb, DIM, 0);

    k_combine<<<N_NODES, 32, 0, stream>>>(zbuf, stb, gtb, nsb, hbuf, ln_g, ln_b, outb, l);
  }

  k_gather<<<g1(BATCH * DIM), 256, 0, stream>>>(outb, (const int*)d_in[IN_P], (float*)d_out);
}
